// MinimalRNNCell_74577812128491
// MI455X (gfx1250) — matrix operand order repaired and checked by reading, not compiled, not run on hardware
//
#include <hip/hip_runtime.h>
#include <hip/hip_bf16.h>
#include <math.h>

#define B_ 64
#define T_ 1024
#define I_ 512
#define H_ 1024
#define O_ 512

typedef __bf16 bf16_t;
typedef __attribute__((ext_vector_type(16))) __bf16 v16bf;
typedef __attribute__((ext_vector_type(8)))  __bf16 v8bf;
typedef __attribute__((ext_vector_type(8)))  float  v8f;

// ---- WMMA fragment loaders (contiguous-k order shared by A and B, wave32) ----
// A-matrix 16x32 bf16: lane L -> m = L&15, g = L>>4.
//   lane holds K = kk+16g .. kk+16g+15 contiguous (same order as B).
static __device__ inline v16bf ld_frag_a(const bf16_t* lane_row, int kk, int g) {
  return *(const v16bf*)(lane_row + kk + 16 * g);
}
// B-matrix 32x16 bf16 (stored as N rows over K, i.e. W[N,K] row-major):
//   lane L -> n = L&15, g = L>>4; lane holds K = kk+16g .. kk+16g+15 contiguous.
static __device__ inline v16bf ld_frag_b(const bf16_t* lane_row, int kk, int g) {
  return *(const v16bf*)(lane_row + kk + 16 * g);
}

static __device__ inline v8f wmma_bf16(v16bf a, v16bf b, v8f c) {
  // (neg_a, A, neg_b, B, c_mod, C, reuse_a, reuse_b)
  return __builtin_amdgcn_wmma_f32_16x16x32_bf16(false, a, false, b, (short)0, c,
                                                 false, false);
}

// 32x64 tile per wave: 2 A frags x 4 B frags -> 8 WMMA per 12 b128 loads.
// acc[2*j+0] covers (m0..m0+15, n0+16j..), acc[2*j+1] covers (m0+16.., n0+16j..)
static __device__ inline void gemm_32x64(const bf16_t* a0p, const bf16_t* a1p,
                                         const bf16_t* b0p, const bf16_t* b1p,
                                         const bf16_t* b2p, const bf16_t* b3p,
                                         int K, int g, v8f acc[8]) {
  for (int kk = 0; kk < K; kk += 32) {
    v16bf a0 = ld_frag_a(a0p, kk, g);
    v16bf a1 = ld_frag_a(a1p, kk, g);
    v16bf b0 = ld_frag_b(b0p, kk, g);
    v16bf b1 = ld_frag_b(b1p, kk, g);
    v16bf b2 = ld_frag_b(b2p, kk, g);
    v16bf b3 = ld_frag_b(b3p, kk, g);
    acc[0] = wmma_bf16(a0, b0, acc[0]);  acc[1] = wmma_bf16(a1, b0, acc[1]);
    acc[2] = wmma_bf16(a0, b1, acc[2]);  acc[3] = wmma_bf16(a1, b1, acc[3]);
    acc[4] = wmma_bf16(a0, b2, acc[4]);  acc[5] = wmma_bf16(a1, b2, acc[5]);
    acc[6] = wmma_bf16(a0, b3, acc[6]);  acc[7] = wmma_bf16(a1, b3, acc[7]);
  }
}

// ---- helpers ----
__global__ void k_cvt(const float* __restrict__ s, bf16_t* __restrict__ d, int n) {
  int i = blockIdx.x * blockDim.x + threadIdx.x;
  if (i < n) d[i] = (bf16_t)s[i];
}

__global__ void k_split_wgate(const float* __restrict__ wg,
                              bf16_t* __restrict__ wgh, bf16_t* __restrict__ wgz) {
  int i = blockIdx.x * blockDim.x + threadIdx.x;   // over H_*H_
  int r = i >> 10, c = i & (H_ - 1);
  wgh[i] = (bf16_t)wg[(size_t)r * (2 * H_) + c];
  wgz[i] = (bf16_t)wg[(size_t)r * (2 * H_) + H_ + c];
}

__global__ void k_copy(const float* __restrict__ s, float* __restrict__ d, int n) {
  int i = blockIdx.x * blockDim.x + threadIdx.x;
  if (i < n) d[i] = s[i];
}

__global__ void k_zero(unsigned* p) { *p = 0u; }

// ---- GEMM 1: z[t*B+b, h] = inputs[b,t,:] . W_in[h,:] + b_in[h]  (bf16 out) ----
__global__ void __launch_bounds__(256) k_gemm_z(const bf16_t* __restrict__ A,
                                                const bf16_t* __restrict__ W,
                                                const float* __restrict__ bias,
                                                bf16_t* __restrict__ Z) {
  int lane = threadIdx.x & 31, wave = threadIdx.x >> 5;
  int tile = blockIdx.x * 8 + wave;
  int mt = tile >> 4, nt = tile & 15;          // N tiles = H/64 = 16
  int m0 = mt * 32, n0 = nt * 64;
  int g = lane >> 4, r = lane & 15;

  int mA0 = m0 + r, mA1 = m0 + 16 + r;
  // output row m = t*B+b maps to input row b*T+t
  const bf16_t* a0p = A + (((size_t)(mA0 & (B_ - 1)) * T_) + (mA0 >> 6)) * I_;
  const bf16_t* a1p = A + (((size_t)(mA1 & (B_ - 1)) * T_) + (mA1 >> 6)) * I_;
  const bf16_t* b0p = W + (size_t)(n0 + r) * I_;
  const bf16_t* b1p = W + (size_t)(n0 + 16 + r) * I_;
  const bf16_t* b2p = W + (size_t)(n0 + 32 + r) * I_;
  const bf16_t* b3p = W + (size_t)(n0 + 48 + r) * I_;

  v8f acc[8] = {};
  gemm_32x64(a0p, a1p, b0p, b1p, b2p, b3p, I_, g, acc);

  auto st = [&](v8f c, int ms, int ns) {
    int n = n0 + ns + r;
    float bv = bias[n];
#pragma unroll
    for (int j = 0; j < 8; ++j) {
      int m = m0 + ms + 8 * g + j;
      Z[(size_t)m * H_ + n] = (bf16_t)(c[j] + bv);
    }
  };
#pragma unroll
  for (int q = 0; q < 4; ++q) {
    st(acc[2 * q + 0], 0, 16 * q);
    st(acc[2 * q + 1], 16, 16 * q);
  }
}

// ---- GEMM 2: a[m, n] = z[m,:] . Wg_z[n,:] + b_gate[n]  (fp32 out) ----
__global__ void __launch_bounds__(256) k_gemm_a(const bf16_t* __restrict__ Zb,
                                                const bf16_t* __restrict__ W,
                                                const float* __restrict__ bias,
                                                float* __restrict__ Aout) {
  int lane = threadIdx.x & 31, wave = threadIdx.x >> 5;
  int tile = blockIdx.x * 8 + wave;
  int mt = tile >> 4, nt = tile & 15;          // N tiles = H/64 = 16
  int m0 = mt * 32, n0 = nt * 64;
  int g = lane >> 4, r = lane & 15;

  const bf16_t* a0p = Zb + (size_t)(m0 + r) * H_;
  const bf16_t* a1p = Zb + (size_t)(m0 + 16 + r) * H_;
  const bf16_t* b0p = W + (size_t)(n0 + r) * H_;
  const bf16_t* b1p = W + (size_t)(n0 + 16 + r) * H_;
  const bf16_t* b2p = W + (size_t)(n0 + 32 + r) * H_;
  const bf16_t* b3p = W + (size_t)(n0 + 48 + r) * H_;

  v8f acc[8] = {};
  gemm_32x64(a0p, a1p, b0p, b1p, b2p, b3p, H_, g, acc);

  auto st = [&](v8f c, int ms, int ns) {
    int n = n0 + ns + r;
    float bv = bias[n];
#pragma unroll
    for (int j = 0; j < 8; ++j) {
      int m = m0 + ms + 8 * g + j;
      Aout[(size_t)m * H_ + n] = c[j] + bv;
    }
  };
#pragma unroll
  for (int q = 0; q < 4; ++q) {
    st(acc[2 * q + 0], 0, 16 * q);
    st(acc[2 * q + 1], 16, 16 * q);
  }
}

// ---- Persistent scan: 8 blocks (co-resident) + software grid barrier ----
__global__ void __launch_bounds__(256) k_scan(const bf16_t* __restrict__ Wgh,
                                              const float* __restrict__ Agate,
                                              const bf16_t* __restrict__ Zbf,
                                              const float* __restrict__ h_in,
                                              const bf16_t* __restrict__ h0bf,
                                              bf16_t* __restrict__ HS,
                                              float* __restrict__ hf0,
                                              float* __restrict__ hf1,
                                              unsigned* __restrict__ bar) {
  int lane = threadIdx.x & 31, wave = threadIdx.x >> 5;
  int wg = blockIdx.x * 8 + wave;       // 0..63 waves
  int mt = wg >> 5, nt = wg & 31;       // 2 x 32 tiles of the 64x1024 state
  int m0 = mt * 32, n0 = nt * 32;
  int g = lane >> 4, r = lane & 15;

  const bf16_t* b0p = Wgh + (size_t)(n0 + r) * H_;
  const bf16_t* b1p = Wgh + (size_t)(n0 + 16 + r) * H_;

  for (int t = 0; t < T_; ++t) {
    const bf16_t* Hp_bf = (t == 0) ? h0bf : (HS + (size_t)(t - 1) * B_ * H_);
    const float*  Hp_f  = (t == 0) ? h_in : ((t & 1) ? hf0 : hf1);
    float*        Hn_f  = (t & 1) ? hf1 : hf0;

    const bf16_t* a0p = Hp_bf + (size_t)(m0 + r) * H_;
    const bf16_t* a1p = Hp_bf + (size_t)(m0 + 16 + r) * H_;

    v8f c00 = {}, c01 = {}, c10 = {}, c11 = {};
    for (int kk = 0; kk < H_; kk += 32) {
      v16bf a0 = ld_frag_a(a0p, kk, g), a1 = ld_frag_a(a1p, kk, g);
      v16bf b0 = ld_frag_b(b0p, kk, g), b1 = ld_frag_b(b1p, kk, g);
      c00 = wmma_bf16(a0, b0, c00);  c01 = wmma_bf16(a0, b1, c01);
      c10 = wmma_bf16(a1, b0, c10);  c11 = wmma_bf16(a1, b1, c11);
    }
    const float*  Ar = Agate + (size_t)t * B_ * H_;
    const bf16_t* Zr = Zbf   + (size_t)t * B_ * H_;
    bf16_t*       Hr = HS    + (size_t)t * B_ * H_;
    auto epi = [&](v8f c, int ms, int ns) {
      int n = n0 + ns + r;
#pragma unroll
      for (int j = 0; j < 8; ++j) {
        int m = m0 + ms + 8 * g + j;
        size_t idx = (size_t)m * H_ + n;
        float pre = c[j] + Ar[idx];
        float u = 1.0f / (1.0f + __expf(-pre));
        float hn = u * Hp_f[idx] + (1.0f - u) * (float)Zr[idx];
        Hn_f[idx] = hn;
        Hr[idx] = (bf16_t)hn;
      }
    };
    epi(c00, 0, 0); epi(c01, 0, 16); epi(c10, 16, 0); epi(c11, 16, 16);

    // grid-wide barrier (monotonic phase counter; 8 blocks co-resident)
    __syncthreads();
    if (threadIdx.x == 0) {
      __threadfence();
      atomicAdd(bar, 1u);
      unsigned target = (unsigned)(t + 1) * gridDim.x;
      while (__hip_atomic_load(bar, __ATOMIC_ACQUIRE, __HIP_MEMORY_SCOPE_AGENT) < target) {
        __builtin_amdgcn_s_sleep(1);
      }
    }
    __syncthreads();
  }
}

// ---- GEMM 3: outs[b,t,o] = hs[t*B+b,:] . W_out[o,:] + b_out[o] ----
__global__ void __launch_bounds__(256) k_gemm_out(const bf16_t* __restrict__ HS,
                                                  const bf16_t* __restrict__ W,
                                                  const float* __restrict__ bias,
                                                  float* __restrict__ Out) {
  int lane = threadIdx.x & 31, wave = threadIdx.x >> 5;
  int tile = blockIdx.x * 8 + wave;
  int mt = tile >> 3, nt = tile & 7;           // N tiles = O/64 = 8
  int m0 = mt * 32, n0 = nt * 64;
  int g = lane >> 4, r = lane & 15;

  const bf16_t* a0p = HS + (size_t)(m0 + r) * H_;
  const bf16_t* a1p = HS + (size_t)(m0 + 16 + r) * H_;
  const bf16_t* b0p = W + (size_t)(n0 + r) * H_;
  const bf16_t* b1p = W + (size_t)(n0 + 16 + r) * H_;
  const bf16_t* b2p = W + (size_t)(n0 + 32 + r) * H_;
  const bf16_t* b3p = W + (size_t)(n0 + 48 + r) * H_;

  v8f acc[8] = {};
  gemm_32x64(a0p, a1p, b0p, b1p, b2p, b3p, H_, g, acc);

  auto st = [&](v8f c, int ms, int ns) {
    int n = n0 + ns + r;
    float bv = bias[n];
#pragma unroll
    for (int j = 0; j < 8; ++j) {
      int m = m0 + ms + 8 * g + j;                       // m = t*B + b
      size_t orow = (size_t)(m & (B_ - 1)) * T_ + (m >> 6);
      Out[orow * O_ + n] = c[j] + bv;
    }
  };
#pragma unroll
  for (int q = 0; q < 4; ++q) {
    st(acc[2 * q + 0], 0, 16 * q);
    st(acc[2 * q + 1], 16, 16 * q);
  }
}

extern "C" void kernel_launch(void* const* d_in, const int* in_sizes, int n_in,
                              void* d_out, int out_size, void* d_ws, size_t ws_size,
                              hipStream_t stream) {
  (void)in_sizes; (void)n_in; (void)out_size; (void)ws_size;
  const float* inputs = (const float*)d_in[0];
  const float* h      = (const float*)d_in[1];
  const float* W_in   = (const float*)d_in[2];
  const float* b_in   = (const float*)d_in[3];
  const float* W_gate = (const float*)d_in[4];
  const float* b_gate = (const float*)d_in[5];
  const float* W_out  = (const float*)d_in[6];
  const float* b_out  = (const float*)d_in[7];
  float* out = (float*)d_out;

  char* ws = (char*)d_ws;
  size_t off = 0;
  auto alloc = [&](size_t bytes) {
    void* p = ws + off;
    off = (off + bytes + 255) & ~(size_t)255;
    return p;
  };
  bf16_t* in_bf   = (bf16_t*)alloc((size_t)B_ * T_ * I_ * 2);
  bf16_t* win_bf  = (bf16_t*)alloc((size_t)H_ * I_ * 2);
  bf16_t* wgh_bf  = (bf16_t*)alloc((size_t)H_ * H_ * 2);
  bf16_t* wgz_bf  = (bf16_t*)alloc((size_t)H_ * H_ * 2);
  bf16_t* wout_bf = (bf16_t*)alloc((size_t)O_ * H_ * 2);
  bf16_t* h0_bf   = (bf16_t*)alloc((size_t)B_ * H_ * 2);
  bf16_t* z_bf    = (bf16_t*)alloc((size_t)T_ * B_ * H_ * 2);
  float*  a_f     = (float*) alloc((size_t)T_ * B_ * H_ * 4);
  bf16_t* hs_bf   = (bf16_t*)alloc((size_t)T_ * B_ * H_ * 2);
  float*  hf0     = (float*) alloc((size_t)B_ * H_ * 4);
  float*  hf1     = (float*) alloc((size_t)B_ * H_ * 4);
  unsigned* bar   = (unsigned*)alloc(256);

  // precision prep (fp32 -> bf16 operands)
  k_cvt<<<(B_ * T_ * I_ + 255) / 256, 256, 0, stream>>>(inputs, in_bf, B_ * T_ * I_);
  k_cvt<<<(H_ * I_ + 255) / 256, 256, 0, stream>>>(W_in, win_bf, H_ * I_);
  k_split_wgate<<<(H_ * H_) / 256, 256, 0, stream>>>(W_gate, wgh_bf, wgz_bf);
  k_cvt<<<(O_ * H_ + 255) / 256, 256, 0, stream>>>(W_out, wout_bf, O_ * H_);
  k_cvt<<<(B_ * H_ + 255) / 256, 256, 0, stream>>>(h, h0_bf, B_ * H_);
  k_zero<<<1, 1, 0, stream>>>(bar);

  // z = inputs @ W_in^T + b_in : M=65536, N=1024, 32x64 tiles, 8 waves/block
  k_gemm_z<<<4096, 256, 0, stream>>>(in_bf, win_bf, b_in, z_bf);
  // a = z @ Wg_z^T + b_gate (hoisted out of the scan): M=65536, N=1024
  k_gemm_a<<<4096, 256, 0, stream>>>(z_bf, wgz_bf, b_gate, a_f);
  // sequential recurrence, persistent kernel, 8 blocks
  k_scan<<<8, 256, 0, stream>>>(wgh_bf, a_f, z_bf, h, h0_bf, hs_bf, hf0, hf1, bar);
  // outs = hs @ W_out^T + b_out : M=65536, N=512
  k_gemm_out<<<2048, 256, 0, stream>>>(hs_bf, wout_bf, b_out, out);
  // h_final (last step wrote hf1 since (T-1) is odd)
  k_copy<<<(B_ * H_ + 255) / 256, 256, 0, stream>>>(hf1, out + (size_t)B_ * T_ * O_,
                                                    B_ * H_);
}
